// VoteAggregation_6923487281653
// MI455X (gfx1250) — compile-verified, matrix operand order checked by reading, not run on hardware
//
#include <hip/hip_runtime.h>
#include <hip/hip_bf16.h>
#include <stdint.h>

// ---------------------------------------------------------------------------
// VoteAggregation for MI455X (gfx1250, wave32).
//  - FPS: 1 block/batch, packed-u64 argmax (ties -> smallest index)
//  - Ball query: 16 rounds of packed-u64 argmin (stable-argsort tie semantics)
//  - GEMMs: v_wmma_f32_16x16x32_f16, KDIM templated (fully unrolled K loop),
//    B pre-packed in fragment-major layout (contiguous 32B per lane ->
//    global_load_b128 pairs, no v_mov_b16 stitching), 4 N-tiles per wave
//    sharing one A fragment.
//  - BatchNorm (training mode): conv bias cancels exactly -> skipped.
// ---------------------------------------------------------------------------

typedef _Float16 half_t;
typedef __attribute__((ext_vector_type(16))) _Float16 v16h;
typedef __attribute__((ext_vector_type(8)))  _Float16 v8h;
typedef __attribute__((ext_vector_type(8)))  float    v8f;

#define BB   8
#define NN   32768
#define PP   256
#define KK   16
#define FF   128
#define CC   256
#define KP1  160            // 130 padded to multiple of 32
#define MM   (BB * PP * KK) // 32768 GEMM rows
#define RADIUS_F 0.3f
#define EPS_F    1e-5f

// ---------------- FPS: one block (1024 threads) per batch --------------------
__global__ void fps_kernel(const float* __restrict__ xyz,
                           float* __restrict__ new_xyz,   // d_out[0 .. B*P*2)
                           float* __restrict__ dist_ws) {
    const int b   = blockIdx.x;
    const int tid = threadIdx.x;
    const float* x = xyz + (size_t)b * NN * 2;
    float* dist    = dist_ws + (size_t)b * NN;

    __shared__ unsigned long long sred[1024];

    for (int i = tid; i < NN; i += 1024) dist[i] = 1e10f;
    __syncthreads();

    int far = 0;
    for (int it = 0; it < PP; ++it) {
        if (tid == 0) {
            new_xyz[((size_t)b * PP + it) * 2 + 0] = x[(size_t)far * 2 + 0];
            new_xyz[((size_t)b * PP + it) * 2 + 1] = x[(size_t)far * 2 + 1];
        }
        const float cx = x[(size_t)far * 2 + 0];
        const float cy = x[(size_t)far * 2 + 1];

        unsigned long long best = 0ull;
        for (int i = tid; i < NN; i += 1024) {
            float dx = x[(size_t)i * 2 + 0] - cx;
            float dy = x[(size_t)i * 2 + 1] - cy;
            float d  = dx * dx + dy * dy;
            float dd = fminf(dist[i], d);
            dist[i]  = dd;
            // argmax with smallest-index tie break: (dist_bits, ~i) maximized
            unsigned long long key =
                ((unsigned long long)__float_as_uint(dd) << 32) | (uint32_t)(~(uint32_t)i);
            best = (key > best) ? key : best;
        }
        sred[tid] = best;
        __syncthreads();
        for (int s = 512; s > 0; s >>= 1) {
            if (tid < s) {
                unsigned long long o = sred[tid + s];
                if (o > sred[tid]) sred[tid] = o;
            }
            __syncthreads();
        }
        far = (int)(~(uint32_t)sred[0]);
        __syncthreads();   // protect sred before next iteration overwrites it
    }
}

// ------------- Ball query: one block (256 threads) per (b,p) -----------------
__global__ void ballquery_kernel(const float* __restrict__ xyz,
                                 const float* __restrict__ new_xyz,
                                 int* __restrict__ group_idx) {
    const int bp  = blockIdx.x;          // b*PP + p
    const int b   = bp / PP;
    const int tid = threadIdx.x;
    const float qx = new_xyz[(size_t)bp * 2 + 0];
    const float qy = new_xyz[(size_t)bp * 2 + 1];
    const float* x = xyz + (size_t)b * NN * 2;

    __shared__ int sel[KK];
    __shared__ unsigned long long sred[256];

    for (int k = 0; k < KK; ++k) {
        unsigned long long best = ~0ull;
        for (int i = tid; i < NN; i += 256) {
            float dx = x[(size_t)i * 2 + 0] - qx;
            float dy = x[(size_t)i * 2 + 1] - qy;
            float d  = dx * dx + dy * dy;
            bool skip = false;
            #pragma unroll
            for (int j = 0; j < KK; ++j)
                if (j < k && sel[j] == i) skip = true;
            if (!skip) {
                // argmin with smallest-index tie break (stable argsort semantics)
                unsigned long long key =
                    ((unsigned long long)__float_as_uint(d) << 32) | (uint32_t)i;
                best = (key < best) ? key : best;
            }
        }
        sred[tid] = best;
        __syncthreads();
        for (int s = 128; s > 0; s >>= 1) {
            if (tid < s) {
                unsigned long long o = sred[tid + s];
                if (o < sred[tid]) sred[tid] = o;
            }
            __syncthreads();
        }
        unsigned long long w = sred[0];
        if (tid == 0) {
            int   idx = (int)(uint32_t)w;
            float d2  = __uint_as_float((uint32_t)(w >> 32));
            sel[k] = idx;
            group_idx[(size_t)bp * KK + k] = (sqrtf(d2) > RADIUS_F) ? 0 : idx;
        }
        __syncthreads();
    }
}

// --------- Build combined A-matrix (M x KP1, f16, zero padded) ---------------
__global__ void build_combined_kernel(const float* __restrict__ xyz,
                                      const float* __restrict__ feats,
                                      const float* __restrict__ new_xyz,
                                      const int* __restrict__ group_idx,
                                      half_t* __restrict__ A) {
    size_t e = (size_t)blockIdx.x * blockDim.x + threadIdx.x;
    if (e >= (size_t)MM * KP1) return;
    int    col = (int)(e % KP1);
    size_t row = e / KP1;                 // = bp*KK + k
    size_t bp  = row / KK;
    int    b   = (int)(bp / PP);
    int    gi  = group_idx[row];
    float v;
    if (col < 2) {
        v = xyz[((size_t)b * NN + gi) * 2 + col] - new_xyz[bp * 2 + col];
    } else if (col < 2 + FF) {
        v = feats[((size_t)b * NN + gi) * FF + (col - 2)];
    } else {
        v = 0.0f;
    }
    A[e] = (half_t)v;
}

// ---- Weight prep: pack into fragment-major B layout -------------------------
// Bp[ ((tn*(Kpad/32) + kb)*32 + lane)*16 + j ]  holds element B[k][n] with
//   n = tn*16 + (lane&15),  h = lane>>4,
//   k = kb*32 + (j<8 ? 8*h + j : 16 + 8*h + (j-8))
// so each lane's 16 halves for one k-block are contiguous (32 bytes).
__global__ void prep_w_pack_kernel(const float* __restrict__ W, half_t* __restrict__ Bp,
                                   int Kpad, int Kreal, int ldw) {
    int e = blockIdx.x * blockDim.x + threadIdx.x;
    if (e >= Kpad * CC) return;
    int j    = e & 15;
    int lane = (e >> 4) & 31;
    int rest = e >> 9;                 // tn * (Kpad/32) + kb
    int kblocks = Kpad >> 5;
    int tn = rest / kblocks;
    int kb = rest - tn * kblocks;
    int h  = lane >> 4;
    int n  = tn * 16 + (lane & 15);
    int k  = kb * 32 + ((j < 8) ? (8 * h + j) : (16 + 8 * h + (j - 8)));
    Bp[e] = (half_t)((k < Kreal) ? W[(size_t)n * ldw + k] : 0.0f);
}

// --------------------- WMMA GEMM: D(M x 256) = A(M x KDIM) * B ---------------
// One wave computes a 16x64 output strip: 4 N-tiles sharing one A fragment.
// KDIM is compile-time -> K loop fully unrolled. B is fragment-major packed.
template<int KDIM>
__global__ void gemm_wmma_kernel(const half_t* __restrict__ A,
                                 const half_t* __restrict__ Bp,
                                 float* __restrict__ D) {
    constexpr int N  = CC;
    constexpr int KB = KDIM / 32;
    const int gtid = blockIdx.x * blockDim.x + threadIdx.x;
    const int wave = gtid >> 5;
    const int lane = threadIdx.x & 31;
    const int tm = wave >> 2;           // row tile
    const int tg = wave & 3;            // group of 4 consecutive N-tiles
    const int hs = lane >> 4;
    const int rc = lane & 15;

    const half_t* Arow = A + (size_t)(tm * 16 + rc) * KDIM;

    v8f acc[4] = {};
    #pragma unroll
    for (int kt = 0; kt < KB; ++kt) {
        const int kb = kt * 32 + 8 * hs;
        v8h alo = *(const v8h*)(Arow + kb);
        v8h ahi = *(const v8h*)(Arow + kb + 16);
        v16h a = __builtin_shufflevector(alo, ahi,
                                         0, 1, 2, 3, 4, 5, 6, 7,
                                         8, 9, 10, 11, 12, 13, 14, 15);
        #pragma unroll
        for (int t = 0; t < 4; ++t) {
            const int tn = tg * 4 + t;
            const v16h bfr =
                *(const v16h*)(Bp + (((size_t)tn * KB + kt) * 32 + lane) * 16);
            acc[t] = __builtin_amdgcn_wmma_f32_16x16x32_f16(
                /*neg_a=*/false, a, /*neg_b=*/false, bfr,
                /*c_mod=*/(short)0, acc[t], /*reuse_a=*/false, /*reuse_b=*/false);
        }
    }
    #pragma unroll
    for (int t = 0; t < 4; ++t) {
        const int ncol = (tg * 4 + t) * 16 + rc;
        #pragma unroll
        for (int r = 0; r < 8; ++r) {
            const int m = tm * 16 + r + 8 * hs;
            D[(size_t)m * N + ncol] = acc[t][r];
        }
    }
}

// ---------- BN stats: one block per channel; scale/shift for (x-mean)*s+beta --
__global__ void bn_stats_kernel(const float* __restrict__ X,
                                const float* __restrict__ gamma,
                                const float* __restrict__ beta,
                                float* __restrict__ scale,
                                float* __restrict__ shift) {
    const int c   = blockIdx.x;
    const int tid = threadIdx.x;
    __shared__ float ssum[256];
    __shared__ float ssq[256];
    float s = 0.f, q = 0.f;
    for (int m = tid; m < MM; m += 256) {
        float v = X[(size_t)m * CC + c];
        s += v; q += v * v;
    }
    ssum[tid] = s; ssq[tid] = q;
    __syncthreads();
    for (int st = 128; st > 0; st >>= 1) {
        if (tid < st) { ssum[tid] += ssum[tid + st]; ssq[tid] += ssq[tid + st]; }
        __syncthreads();
    }
    if (tid == 0) {
        float mean = ssum[0] / (float)MM;
        float var  = ssq[0] / (float)MM - mean * mean;
        float sc   = gamma[c] * rsqrtf(var + EPS_F);
        scale[c] = sc;
        shift[c] = beta[c] - mean * sc;
    }
}

// ---------- Apply BN+ReLU, emit f16 for next GEMM ----------------------------
__global__ void bn_apply_f16_kernel(const float* __restrict__ X,
                                    const float* __restrict__ scale,
                                    const float* __restrict__ shift,
                                    half_t* __restrict__ Y) {
    size_t e = (size_t)blockIdx.x * blockDim.x + threadIdx.x;
    if (e >= (size_t)MM * CC) return;
    int c = (int)(e % CC);
    float y = fmaxf(X[e] * scale[c] + shift[c], 0.0f);
    Y[e] = (half_t)y;
}

// ---------- Apply BN+ReLU + max-pool over K, write pooled output -------------
__global__ void bn_apply_pool_kernel(const float* __restrict__ X,
                                     const float* __restrict__ scale,
                                     const float* __restrict__ shift,
                                     float* __restrict__ pooled) {
    const int bp = blockIdx.x;       // b*PP + p
    const int c  = threadIdx.x;      // 256 channels
    float sc = scale[c], sh = shift[c];
    float best = 0.0f;               // relu(x) >= 0, so 0 is a safe identity
    #pragma unroll
    for (int k = 0; k < KK; ++k) {
        float y = fmaxf(X[((size_t)bp * KK + k) * CC + c] * sc + sh, 0.0f);
        best = fmaxf(best, y);
    }
    pooled[(size_t)bp * CC + c] = best;
}

// ---------------------------------------------------------------------------
extern "C" void kernel_launch(void* const* d_in, const int* in_sizes, int n_in,
                              void* d_out, int out_size, void* d_ws, size_t ws_size,
                              hipStream_t stream) {
    const float* votes_xyz   = (const float*)d_in[0];   // (B,N,2)
    const float* votes_feats = (const float*)d_in[1];   // (B,N,F)
    const float* W1          = (const float*)d_in[2];   // (C, F+2)
    // d_in[3] = b1 : cancels exactly under training-mode BatchNorm -> unused
    const float* g1          = (const float*)d_in[4];
    const float* be1         = (const float*)d_in[5];
    const float* W2          = (const float*)d_in[6];   // (C, C)
    // d_in[7] = b2 : unused (same reason)
    const float* g2          = (const float*)d_in[8];
    const float* be2         = (const float*)d_in[9];

    float* out_newxyz = (float*)d_out;                       // B*P*2 = 4096 floats
    float* out_pooled = (float*)d_out + (size_t)BB * PP * 2; // B*P*C floats

    // Workspace carve-up (all offsets 256B aligned; total ~62.3 MB)
    char* ws = (char*)d_ws;
    float*  dist_ws = (float*) (ws + 0);                     //  1,048,576 B
    int*    gidx    = (int*)   (ws + 1048576);               //    131,072 B
    half_t* combA   = (half_t*)(ws + 1179648);               // 10,485,760 B
    float*  bnbuf   = (float*) (ws + 11665408);              //      4,096 B
    float*  scale1  = bnbuf,       *shift1 = bnbuf + 256;
    float*  scale2  = bnbuf + 512, *shift2 = bnbuf + 768;
    half_t* Bm1     = (half_t*)(ws + 11669504);              //     81,920 B
    half_t* Bm2     = (half_t*)(ws + 11751424);              //    131,072 B
    float*  h1f32   = (float*) (ws + 11882496);              // 33,554,432 B (reused as h2f32)
    half_t* h1f16   = (half_t*)(ws + 45436928);              // 16,777,216 B
    float*  h2f32   = h1f32;   // h1f32 dead once h1f16 is built

    // 1) FPS (serial chain; 1 block per batch)
    fps_kernel<<<BB, 1024, 0, stream>>>(votes_xyz, out_newxyz, dist_ws);

    // 2) Ball query: 16-NN with radius clamp-to-index-0
    ballquery_kernel<<<BB * PP, 256, 0, stream>>>(votes_xyz, out_newxyz, gidx);

    // 3) Weight prep (f16, fragment-major packed B layout)
    prep_w_pack_kernel<<<(KP1 * CC + 255) / 256, 256, 0, stream>>>(W1, Bm1, KP1, FF + 2, FF + 2);
    prep_w_pack_kernel<<<(CC * CC + 255) / 256, 256, 0, stream>>>(W2, Bm2, CC, CC, CC);

    // 4) Gather + normalize + concat -> A matrix (M x 160, f16)
    {
        size_t tot = (size_t)MM * KP1;
        build_combined_kernel<<<(unsigned)((tot + 255) / 256), 256, 0, stream>>>(
            votes_xyz, votes_feats, out_newxyz, gidx, combA);
    }

    // 5) GEMM1: (32768 x 160) * (160 x 256) -> h1f32  [v_wmma_f32_16x16x32_f16]
    {
        int waves = (MM / 16) * 4;                    // 4 N-tiles per wave
        gemm_wmma_kernel<KP1><<<waves * 32 / 256, 256, 0, stream>>>(combA, Bm1, h1f32);
    }

    // 6) BN1 stats + apply (ReLU, to f16)
    bn_stats_kernel<<<CC, 256, 0, stream>>>(h1f32, g1, be1, scale1, shift1);
    bn_apply_f16_kernel<<<(unsigned)(((size_t)MM * CC + 255) / 256), 256, 0, stream>>>(
        h1f32, scale1, shift1, h1f16);

    // 7) GEMM2: (32768 x 256) * (256 x 256) -> h2f32
    {
        int waves = (MM / 16) * 4;
        gemm_wmma_kernel<CC><<<waves * 32 / 256, 256, 0, stream>>>(h1f16, Bm2, h2f32);
    }

    // 8) BN2 stats + apply + max-pool over K -> pooled output
    bn_stats_kernel<<<CC, 256, 0, stream>>>(h2f32, g2, be2, scale2, shift2);
    bn_apply_pool_kernel<<<BB * PP, CC, 0, stream>>>(h2f32, scale2, shift2, out_pooled);
}
